// MultiHeadAttention_74663711473860
// MI455X (gfx1250) — compile-verified, matrix operand order checked by reading, not run on hardware
//
#include <hip/hip_runtime.h>

// Problem constants (compile-time; match reference).
#define D_MODEL   1024
#define NUM_HEADS 16
#define D_HEAD    64
#define BATCH     2
#define SEQ       2048
#define HB        (NUM_HEADS * BATCH)   // 32
#define SPITCH    (SEQ + 4)             // LDS score row pitch: rows 4 banks apart

typedef __attribute__((ext_vector_type(16))) _Float16 v16h;
typedef __attribute__((ext_vector_type(8)))  _Float16 v8h;
typedef __attribute__((ext_vector_type(8)))  float    v8f;
typedef __attribute__((ext_vector_type(4)))  float    f32x4;

// ---------------------------------------------------------------------------
// Fragment builders. CDNA5 16-bit A/B layout (ISA 7.12.2):
//   lane (0..15): row/col = lane, K-phase 0;  lane (16..31): same row/col, +8
//   v16h elements 0..7  -> K = kb+0..kb+7      (kb = (lane>>4)*8)
//   v16h elements 8..15 -> K = kb+16..kb+23
// Caller passes p = row_base + k0 + kb.
// ---------------------------------------------------------------------------
static __device__ __forceinline__ v16h frag_from_f32(const float* p) {
  const f32x4 a0 = *reinterpret_cast<const f32x4*>(p);
  const f32x4 a1 = *reinterpret_cast<const f32x4*>(p + 4);
  const f32x4 a2 = *reinterpret_cast<const f32x4*>(p + 16);
  const f32x4 a3 = *reinterpret_cast<const f32x4*>(p + 20);
  v16h f;
#pragma unroll
  for (int i = 0; i < 4; ++i) {
    f[i]      = (_Float16)a0[i];
    f[i + 4]  = (_Float16)a1[i];
    f[i + 8]  = (_Float16)a2[i];
    f[i + 12] = (_Float16)a3[i];
  }
  return f;
}

static __device__ __forceinline__ v16h frag_from_f16(const _Float16* p) {
  const v8h lo = *reinterpret_cast<const v8h*>(p);
  const v8h hi = *reinterpret_cast<const v8h*>(p + 16);
  v16h f;
#pragma unroll
  for (int i = 0; i < 8; ++i) { f[i] = lo[i]; f[i + 8] = hi[i]; }
  return f;
}

static __device__ __forceinline__ v8f wmma16(v16h a, v16h b, v8f c) {
  return __builtin_amdgcn_wmma_f32_16x16x32_f16(false, a, false, b, (short)0, c,
                                                false, false);
}

// LDS byte offset of a generic pointer into a __shared__ array: the LDS
// aperture maps addr[31:0] directly onto the LDS allocation (ISA 10.2).
static __device__ __forceinline__ unsigned lds_off(const void* p) {
  return (unsigned)(size_t)p;
}

// ---------------------------------------------------------------------------
// Kernel 0: bulk f32 -> f16 conversion (streams X and W once; removes all
// cvt work from GEMM inner loops and halves their load bytes).
// ---------------------------------------------------------------------------
__global__ void __launch_bounds__(256)
cvt_f16_kernel(const float* __restrict__ src, _Float16* __restrict__ dst) {
  const size_t i = ((size_t)blockIdx.x * 256 + threadIdx.x) * 8;
  const f32x4 a = *reinterpret_cast<const f32x4*>(src + i);
  const f32x4 b = *reinterpret_cast<const f32x4*>(src + i + 4);
  v8h o;
#pragma unroll
  for (int j = 0; j < 4; ++j) { o[j] = (_Float16)a[j]; o[j + 4] = (_Float16)b[j]; }
  *reinterpret_cast<v8h*>(dst + i) = o;
}

// ---------------------------------------------------------------------------
// Kernel 1: fused QKV projection, f16 operands. True ping-pong pipeline:
// two fragment buffers alternate between load and WMMA use with no rotation
// copies; loads for chunk k+64 fly while chunk k+32 feeds the matrix pipe.
// q16/k16 stored [hb][s][Dh] f16; v stored transposed vT[hb][d][s].
// ---------------------------------------------------------------------------
__global__ void __launch_bounds__(128)
qkv_proj_kernel(const _Float16* __restrict__ X0, const _Float16* __restrict__ X1,
                const _Float16* __restrict__ X2,
                const _Float16* __restrict__ W0, const _Float16* __restrict__ W1,
                const _Float16* __restrict__ W2,
                const float* __restrict__ B0, const float* __restrict__ B1,
                const float* __restrict__ B2,
                _Float16* __restrict__ q16, _Float16* __restrict__ k16,
                _Float16* __restrict__ vT) {
  const int lane = threadIdx.x & 31;
  const int wid  = threadIdx.x >> 5;
  const int gid  = blockIdx.x * 4 + wid;          // 3*256*16 strips total
  const int mat  = gid / (256 * 16);
  const int rem  = gid % (256 * 16);
  const int mtile = rem >> 4;                     // 0..255  (16 rows)
  const int n0    = (rem & 15) << 6;              // 0..960  (64 cols)

  const _Float16* X    = (mat == 0) ? X0 : (mat == 1) ? X1 : X2;
  const _Float16* W    = (mat == 0) ? W0 : (mat == 1) ? W1 : W2;
  const float*    bias = (mat == 0) ? B0 : (mat == 1) ? B1 : B2;

  const int ln = lane & 15;
  const int kb = (lane >> 4) << 3;
  const _Float16* xrow = X + (size_t)(mtile * 16 + ln) * D_MODEL + kb;
  const _Float16* wr[4];
#pragma unroll
  for (int nn = 0; nn < 4; ++nn)
    wr[nn] = W + (size_t)(n0 + nn * 16 + ln) * D_MODEL + kb;

  v8f acc[4] = {v8f{}, v8f{}, v8f{}, v8f{}};

  // Ping-pong buffers; buffer 0 preloaded with chunk 0.
  v16h a0 = frag_from_f16(xrow);
  v16h b0[4];
#pragma unroll
  for (int nn = 0; nn < 4; ++nn) b0[nn] = frag_from_f16(wr[nn]);

  for (int k0 = 0; k0 < D_MODEL - 64; k0 += 64) {
    __builtin_prefetch(xrow + k0 + 256, 0, 1);
    // load buffer 1 = chunk k0+32
    const v16h a1 = frag_from_f16(xrow + k0 + 32);
    v16h b1[4];
#pragma unroll
    for (int nn = 0; nn < 4; ++nn) b1[nn] = frag_from_f16(wr[nn] + k0 + 32);
    // compute chunk k0 from buffer 0
#pragma unroll
    for (int nn = 0; nn < 4; ++nn) acc[nn] = wmma16(a0, b0[nn], acc[nn]);
    // reload buffer 0 = chunk k0+64
    a0 = frag_from_f16(xrow + k0 + 64);
#pragma unroll
    for (int nn = 0; nn < 4; ++nn) b0[nn] = frag_from_f16(wr[nn] + k0 + 64);
    // compute chunk k0+32 from buffer 1
#pragma unroll
    for (int nn = 0; nn < 4; ++nn) acc[nn] = wmma16(a1, b1[nn], acc[nn]);
  }
  // Tail: buffer 0 holds chunk D_MODEL-64; load and compute the last two.
  {
    const v16h a1 = frag_from_f16(xrow + D_MODEL - 32);
    v16h b1[4];
#pragma unroll
    for (int nn = 0; nn < 4; ++nn) b1[nn] = frag_from_f16(wr[nn] + D_MODEL - 32);
#pragma unroll
    for (int nn = 0; nn < 4; ++nn) acc[nn] = wmma16(a0, b0[nn], acc[nn]);
#pragma unroll
    for (int nn = 0; nn < 4; ++nn) acc[nn] = wmma16(a1, b1[nn], acc[nn]);
  }

  // C layout: VGPR r holds M = kb + r, N = ln.
  const int mbase = mtile * 16 + kb;
  const int bidx  = mbase >> 11;                  // batch
  const int sbase = mbase & (SEQ - 1);
#pragma unroll
  for (int nn = 0; nn < 4; ++nn) {
    const int   ncol = n0 + nn * 16 + ln;
    const float bv   = bias[ncol];
    const int   h    = ncol >> 6;
    const int   d    = ncol & 63;
    const int   hbi  = h * BATCH + bidx;
    if (mat < 2) {
      _Float16* dst = (mat == 0) ? q16 : k16;
#pragma unroll
      for (int r = 0; r < 8; ++r)
        dst[((size_t)hbi * SEQ + (sbase + r)) * D_HEAD + d] =
            (_Float16)(acc[nn][r] + bv);
    } else {
      v8h outv;
#pragma unroll
      for (int r = 0; r < 8; ++r) outv[r] = (_Float16)(acc[nn][r] + bv);
      *reinterpret_cast<v8h*>(vT + (size_t)(hbi * D_HEAD + d) * SEQ + sbase) = outv;
    }
  }
}

// ---------------------------------------------------------------------------
// Kernel 2: fused attention. One 256-thread block (8 waves) owns 16 q-rows of
// one (head*batch): scores -> LDS, softmax in LDS, ASYNC LDS->global DMA of
// the normalized attn tile (overlapped with the context GEMM reading attn
// straight from LDS; split-K over the 8 waves, partials reduced through LDS).
// ---------------------------------------------------------------------------
__global__ void __launch_bounds__(256)
fused_attn_kernel(const _Float16* __restrict__ q16, const _Float16* __restrict__ k16,
                  const _Float16* __restrict__ vT,
                  float* __restrict__ attn_out, float* __restrict__ ctx_out) {
  __shared__ float sS[16 * SPITCH];   // 16 x 2048 scores (padded rows) ~128 KB
  __shared__ float sRed[16 * 16];     // per-row reduction scratch

  const int tid  = threadIdx.x;
  const int lane = tid & 31;
  const int wid  = tid >> 5;
  const int hb   = blockIdx.x >> 7;   // 0..31
  const int qtile = blockIdx.x & 127; // 0..127
  const int ln = lane & 15;
  const int kb = (lane >> 4) << 3;

  // ---- Phase 1: scores = scale * Q K^T into LDS -------------------------
  const _Float16* qrow = q16 + ((size_t)hb * SEQ + qtile * 16 + ln) * D_HEAD;
  const v16h aq0 = frag_from_f16(qrow + 0 + kb);
  const v16h aq1 = frag_from_f16(qrow + 32 + kb);

#pragma unroll
  for (int strip = 0; strip < 4; ++strip) {
    const int n0 = wid * 256 + strip * 64;        // 64 key columns
    v8f acc[4] = {v8f{}, v8f{}, v8f{}, v8f{}};
    const _Float16* kr[4];
#pragma unroll
    for (int nn = 0; nn < 4; ++nn)
      kr[nn] = k16 + ((size_t)hb * SEQ + n0 + nn * 16 + ln) * D_HEAD;
#pragma unroll
    for (int nn = 0; nn < 4; ++nn)
      acc[nn] = wmma16(aq0, frag_from_f16(kr[nn] + kb), acc[nn]);
#pragma unroll
    for (int nn = 0; nn < 4; ++nn)
      acc[nn] = wmma16(aq1, frag_from_f16(kr[nn] + 32 + kb), acc[nn]);
#pragma unroll
    for (int nn = 0; nn < 4; ++nn) {
      const int col = n0 + nn * 16 + ln;
#pragma unroll
      for (int r = 0; r < 8; ++r)
        sS[(kb + r) * SPITCH + col] = acc[nn][r] * 0.125f;  // 1/sqrt(Dh)
    }
  }
  __syncthreads();

  // ---- Phase 2: softmax over each of the 16 rows ------------------------
  const int row = tid >> 4;                       // 0..15
  const int sub = tid & 15;                       // 0..15 (128 cols each)
  float* rp = sS + row * SPITCH + sub * 128;

  float lm = -3.402823466e+38f;
#pragma unroll 8
  for (int i = 0; i < 128; ++i) lm = fmaxf(lm, rp[i]);
  sRed[row * 16 + sub] = lm;
  __syncthreads();
  for (int off = 8; off > 0; off >>= 1) {
    if (sub < off)
      sRed[row * 16 + sub] = fmaxf(sRed[row * 16 + sub], sRed[row * 16 + sub + off]);
    __syncthreads();
  }
  const float rmax = sRed[row * 16];
  __syncthreads();

  float ls = 0.0f;
#pragma unroll 8
  for (int i = 0; i < 128; ++i) {
    const float e = __expf(rp[i] - rmax);
    rp[i] = e;
    ls += e;
  }
  sRed[row * 16 + sub] = ls;
  __syncthreads();
  for (int off = 8; off > 0; off >>= 1) {
    if (sub < off)
      sRed[row * 16 + sub] += sRed[row * 16 + sub + off];
    __syncthreads();
  }
  const float inv = 1.0f / sRed[row * 16];
#pragma unroll 8
  for (int i = 0; i < 128; ++i) rp[i] *= inv;
  __syncthreads();

  // ---- Kick off async LDS -> global DMA of the attn tile (NT stream) ----
  // GLOBAL_STORE_ASYNC_FROM_LDS_B128: per lane, 16 bytes LDS -> memory,
  // tracked by ASYNCcnt; overlaps with the phase-3 WMMA work below.
  {
    float* gA = attn_out + (size_t)hb * SEQ * SEQ + (size_t)(qtile * 16) * SEQ;
#pragma unroll
    for (int r = 0; r < 16; ++r) {
#pragma unroll
      for (int pass = 0; pass < 2; ++pass) {
        const int c4 = (pass * 256 + tid) * 4;            // float column
        const unsigned lsrc = lds_off(sS + r * SPITCH + c4);
        float* gdst = gA + (size_t)r * SEQ + c4;
        asm volatile(
            "global_store_async_from_lds_b128 %0, %1, off th:TH_STORE_NT"
            :: "v"(gdst), "v"(lsrc) : "memory");
      }
    }
  }

  // ---- Phase 3: context = attn @ V (split-K across the 8 waves) ---------
  v8f cacc[4] = {v8f{}, v8f{}, v8f{}, v8f{}};
  const _Float16* vr[4];
#pragma unroll
  for (int nn = 0; nn < 4; ++nn)
    vr[nn] = vT + (size_t)(hb * D_HEAD + nn * 16 + ln) * SEQ + kb;
  const float* arow = sS + ln * SPITCH + kb;
  const int kBeg = wid * 256;

  // Double-buffer the global V fragments (fully unrolled: copies fold away).
  v16h bv[4];
#pragma unroll
  for (int nn = 0; nn < 4; ++nn) bv[nn] = frag_from_f16(vr[nn] + kBeg);
#pragma unroll
  for (int kk = 0; kk < 256; kk += 32) {
    v16h bn[4];
    if (kk + 32 < 256) {
#pragma unroll
      for (int nn = 0; nn < 4; ++nn)
        bn[nn] = frag_from_f16(vr[nn] + kBeg + kk + 32);
    }
    const v16h a = frag_from_f32(arow + kBeg + kk);     // LDS fp32 attn
#pragma unroll
    for (int nn = 0; nn < 4; ++nn) cacc[nn] = wmma16(a, bv[nn], cacc[nn]);
    if (kk + 32 < 256) {
#pragma unroll
      for (int nn = 0; nn < 4; ++nn) bv[nn] = bn[nn];
    }
  }

  // Async DMA must be fully drained before sS is recycled for partials.
  asm volatile("s_wait_asynccnt 0" ::: "memory");
  __syncthreads();

#pragma unroll
  for (int nn = 0; nn < 4; ++nn) {
    const int d = nn * 16 + ln;
#pragma unroll
    for (int r = 0; r < 8; ++r)
      sS[wid * 1024 + (kb + r) * 64 + d] = cacc[nn][r];
  }
  __syncthreads();

  // Reduce 8 partials and scatter into [B, S, H*Dh].
  const int h = hb >> 1;
  const int b = hb & 1;
#pragma unroll
  for (int i = 0; i < 4; ++i) {
    const int idx = i * 256 + tid;    // m*64 + d
    const int m = idx >> 6;
    const int d = idx & 63;
    float sum = 0.0f;
#pragma unroll
    for (int w = 0; w < 8; ++w) sum += sS[w * 1024 + idx];
    __builtin_nontemporal_store(
        sum, ctx_out + (size_t)(b * SEQ + qtile * 16 + m) * D_MODEL + h * D_HEAD + d);
  }
}

// ---------------------------------------------------------------------------
extern "C" void kernel_launch(void* const* d_in, const int* in_sizes, int n_in,
                              void* d_out, int out_size, void* d_ws, size_t ws_size,
                              hipStream_t stream) {
  const float* query = (const float*)d_in[0];
  const float* key   = (const float*)d_in[1];
  const float* value = (const float*)d_in[2];
  const float* Wq    = (const float*)d_in[3];
  const float* bq    = (const float*)d_in[4];
  const float* Wk    = (const float*)d_in[5];
  const float* bk    = (const float*)d_in[6];
  const float* Wv    = (const float*)d_in[7];
  const float* bv    = (const float*)d_in[8];

  float* out  = (float*)d_out;                          // context [B,S,D]
  float* attn = out + (size_t)BATCH * SEQ * D_MODEL;    // attn [HB,S,S]

  const size_t xElems = (size_t)BATCH * SEQ * D_MODEL;  // 4 Mi
  const size_t wElems = (size_t)D_MODEL * D_MODEL;      // 1 Mi
  const size_t pElems = (size_t)HB * SEQ * D_HEAD;      // 4 Mi

  _Float16* Xq16 = (_Float16*)d_ws;
  _Float16* Xk16 = Xq16 + xElems;
  _Float16* Xv16 = Xk16 + xElems;
  _Float16* Wq16 = Xv16 + xElems;
  _Float16* Wk16 = Wq16 + wElems;
  _Float16* Wv16 = Wk16 + wElems;
  _Float16* q16  = Wv16 + wElems;
  _Float16* k16  = q16 + pElems;
  _Float16* vT   = k16 + pElems;

  // 0) f32 -> f16 pre-conversion (2048 elems per block of 256 threads)
  cvt_f16_kernel<<<xElems / 2048, 256, 0, stream>>>(query, Xq16);
  cvt_f16_kernel<<<xElems / 2048, 256, 0, stream>>>(key,   Xk16);
  cvt_f16_kernel<<<xElems / 2048, 256, 0, stream>>>(value, Xv16);
  cvt_f16_kernel<<<wElems / 2048, 256, 0, stream>>>(Wq,    Wq16);
  cvt_f16_kernel<<<wElems / 2048, 256, 0, stream>>>(Wk,    Wk16);
  cvt_f16_kernel<<<wElems / 2048, 256, 0, stream>>>(Wv,    Wv16);

  // 1) QKV projections (3 * 256 * 16 strips, 4 waves/block)
  qkv_proj_kernel<<<(3 * 256 * 16) / 4, 128, 0, stream>>>(
      Xq16, Xk16, Xv16, Wq16, Wk16, Wv16, bq, bk, bv, q16, k16, vT);

  // 2) fused scores + softmax + async attn DMA + context
  fused_attn_kernel<<<HB * 128, 256, 0, stream>>>(q16, k16, vT, attn, out);
}
